// Attention_49005576847523
// MI455X (gfx1250) — matrix operand order repaired and checked by reading, not compiled, not run on hardware
//
#include <hip/hip_runtime.h>
#include <stdint.h>

typedef __attribute__((ext_vector_type(16))) _Float16 v16h;
typedef __attribute__((ext_vector_type(8)))  _Float16 v8h;
typedef __attribute__((ext_vector_type(8)))  float    v8f;
typedef __attribute__((ext_vector_type(4)))  unsigned int v4u;
typedef __attribute__((ext_vector_type(4)))  int      v4i;
typedef __attribute__((ext_vector_type(8)))  int      v8i;

#define WMMA_F16(A, Bm, C) \
  __builtin_amdgcn_wmma_f32_16x16x32_f16(false, (A), false, (Bm), (short)0, (C), false, false)

constexpr int BATCH = 8;
constexpr int SQ    = 2048;
constexpr int SK    = 2048;
constexpr int HD    = 256;   // query/key dim
constexpr int HDV   = 256;   // value dim
constexpr int DIN   = HD + HDV;  // 512 projection input dim
constexpr int BQ    = 128;   // q rows per block (8 waves x 16)
constexpr int BK    = 32;    // keys per main-loop tile / W input chunk
constexpr int NW    = 8;     // waves per block
constexpr int NTHREADS = 256;
constexpr int NKT   = SK / BK;                  // 64 key tiles
constexpr size_t NE = (size_t)BATCH * SK * HD;  // elems per K / V tensor

// ---------------- shared device helpers ----------------

// A-matrix (16x32 f16) fragment from a row-major 16x32 f16 tile in LDS.
// Contiguous-k order used here (B loads match): lane&15 = row M; elems 0..15 = K[hi*16..hi*16+16).
__device__ __forceinline__ v16h loadA32(const _Float16* t, int ln, int hi8) {
  v8h lo = *(const v8h*)&t[ln * 32 + 2 * hi8];
  v8h hh = *(const v8h*)&t[ln * 32 + 2 * hi8 + 8];
  v16h a;
#pragma unroll
  for (int i = 0; i < 8; ++i) { a[i] = lo[i]; a[i + 8] = hh[i]; }
  return a;
}

// A fragment from a row of fp32 global memory (for resident Q fragments).
__device__ __forceinline__ v16h loadAQ(const float* qrow, int f, int hi8) {
  const float* p = qrow + f * 32 + 2 * hi8;
  float4 a0 = *(const float4*)(p);
  float4 a1 = *(const float4*)(p + 4);
  float4 b0 = *(const float4*)(p + 8);
  float4 b1 = *(const float4*)(p + 12);
  v16h a;
  a[0]  = (_Float16)a0.x; a[1]  = (_Float16)a0.y; a[2]  = (_Float16)a0.z; a[3]  = (_Float16)a0.w;
  a[4]  = (_Float16)a1.x; a[5]  = (_Float16)a1.y; a[6]  = (_Float16)a1.z; a[7]  = (_Float16)a1.w;
  a[8]  = (_Float16)b0.x; a[9]  = (_Float16)b0.y; a[10] = (_Float16)b0.z; a[11] = (_Float16)b0.w;
  a[12] = (_Float16)b1.x; a[13] = (_Float16)b1.y; a[14] = (_Float16)b1.z; a[15] = (_Float16)b1.w;
  return a;
}

// TDM: DMA one 2D f16 tile (tl1 rows x tl0 elems, row stride st0) global -> LDS.
// D# packing per CDNA5 ISA 8.3/8.4: count=1, data_size=2B, type=2; groups 2/3 zero (2D).
__device__ __forceinline__ void tdm_load_2d_f16(const _Float16* gsrc, unsigned lds_off,
                                                unsigned td0, unsigned td1,
                                                unsigned tl0, unsigned tl1,
                                                unsigned st0) {
  unsigned long long ga = (unsigned long long)(uintptr_t)gsrc;
  v4u g0;
  g0[0] = 1u;                                   // count=1 (valid user descriptor)
  g0[1] = lds_off;                              // lds_addr (bytes)
  g0[2] = (unsigned)ga;                         // global_addr[31:0]
  g0[3] = (unsigned)((ga >> 32) & 0x1FFFFFFull) | (2u << 30);  // addr[56:32] | type=2
  v8i g1;
  g1[0] = (int)(1u << 16);                      // data_size=1 => 2 bytes
  g1[1] = (int)((td0 & 0xFFFFu) << 16);         // tensor_dim0[15:0] @ bits[63:48]
  g1[2] = (int)((td0 >> 16) | ((td1 & 0xFFFFu) << 16));  // dim0[31:16] | dim1[15:0]
  g1[3] = (int)((td1 >> 16) | (tl0 << 16));     // dim1[31:16] | tile_dim0
  g1[4] = (int)tl1;                             // tile_dim1 (tile_dim2 = 0)
  g1[5] = (int)st0;                             // tensor_dim0_stride[31:0]
  g1[6] = 0;
  g1[7] = 0;
  v4i gz = {0, 0, 0, 0};
#if __clang_major__ >= 23
  v8i gz8 = {0, 0, 0, 0, 0, 0, 0, 0};
  __builtin_amdgcn_tensor_load_to_lds(g0, g1, gz, gz, gz8, 0);
#else
  __builtin_amdgcn_tensor_load_to_lds(g0, g1, gz, gz, 0);
#endif
}

// One 32-key flash-attention step: S = Q*K^T, online softmax, acc += P*V.
__device__ __forceinline__ void attn_tile(const _Float16* __restrict__ sKt,
                                          const _Float16* __restrict__ sVtT,
                                          _Float16* __restrict__ sPw,
                                          const v16h (&qf)[8], v8f (&acc)[16],
                                          float (&m_run)[8], float (&l_run)[8],
                                          int ln, int hi8, int hi16) {
  const v8f zero8 = {0.f, 0.f, 0.f, 0.f, 0.f, 0.f, 0.f, 0.f};
  v8f s0 = zero8, s1 = zero8;
#pragma unroll
  for (int f = 0; f < 8; ++f) {
    v16h b0 = *(const v16h*)&sKt[(0 + ln) * HD + f * 32 + hi16];
    v16h b1 = *(const v16h*)&sKt[(16 + ln) * HD + f * 32 + hi16];
    s0 = WMMA_F16(qf[f], b0, s0);
    s1 = WMMA_F16(qf[f], b1, s1);
  }
  const float scale = 0.0625f;  // 1/sqrt(256)
  float alpha[8];
#pragma unroll
  for (int j = 0; j < 8; ++j) {
    float x0 = s0[j] * scale;
    float x1 = s1[j] * scale;
    float mx = fmaxf(x0, x1);
#pragma unroll
    for (int m = 1; m < 16; m <<= 1) mx = fmaxf(mx, __shfl_xor(mx, m, 32));
    const float mNew = fmaxf(m_run[j], mx);
    alpha[j] = __expf(m_run[j] - mNew);
    const float p0 = __expf(x0 - mNew);
    const float p1 = __expf(x1 - mNew);
    sPw[(j + hi8) * BK + ln]      = (_Float16)p0;   // C-layout -> row-major stage
    sPw[(j + hi8) * BK + 16 + ln] = (_Float16)p1;
    float rs = p0 + p1;
#pragma unroll
    for (int m = 1; m < 16; m <<= 1) rs += __shfl_xor(rs, m, 32);
    l_run[j] = l_run[j] * alpha[j] + rs;
    m_run[j] = mNew;
  }
#pragma unroll
  for (int n = 0; n < 16; ++n) {
#pragma unroll
    for (int j = 0; j < 8; ++j) acc[n][j] *= alpha[j];
  }
  v16h pa = loadA32(sPw, ln, hi8);   // same-wave LDS ops are in-order
#pragma unroll
  for (int n = 0; n < 16; ++n) {
    v16h bv = *(const v16h*)&sVtT[(n * 16 + ln) * BK + hi16];
    acc[n] = WMMA_F16(pa, bv, acc[n]);
  }
}

// Projection epilogue, fallback flavor: W chunks staged from fp32 global by VALU.
__device__ __forceinline__ void fused_projection(v8f (&acc)[16], const v16h (&qf)[8],
                                                 const float (&l_run)[8],
                                                 _Float16* sWt, _Float16* sPw,
                                                 const float* __restrict__ W,
                                                 const float* __restrict__ bias,
                                                 float* __restrict__ ob,
                                                 int tid, int ln, int hi8, int hi16,
                                                 int qRow0) {
  const v8f zero8 = {0.f, 0.f, 0.f, 0.f, 0.f, 0.f, 0.f, 0.f};
  float invl[8];
#pragma unroll
  for (int j = 0; j < 8; ++j) invl[j] = 1.0f / l_run[j];
  v8f oacc[16];
#pragma unroll
  for (int n = 0; n < 16; ++n) oacc[n] = zero8;

  for (int c = 0; c < 16; ++c) {
    __syncthreads();
    for (int idx = tid; idx < BK * HD; idx += NTHREADS) {
      const int il = idx >> 8;    // in_local 0..31
      const int oc = idx & 255;   // out col
      sWt[oc * BK + il] = (_Float16)W[(size_t)(c * 32 + il) * HD + oc];
    }
    __syncthreads();
    v16h a;
    if (c < 8) {
#pragma unroll
      for (int t = 0; t < 2; ++t) {
        const int n = 2 * c + t;
#pragma unroll
        for (int j = 0; j < 8; ++j)
          sPw[(j + hi8) * BK + t * 16 + ln] = (_Float16)(acc[n][j] * invl[j]);
      }
      a = loadA32(sPw, ln, hi8);
    } else {
      a = qf[c - 8];
    }
#pragma unroll
    for (int n = 0; n < 16; ++n) {
      v16h bw = *(const v16h*)&sWt[(n * 16 + ln) * BK + hi16];
      oacc[n] = WMMA_F16(a, bw, oacc[n]);
    }
  }
#pragma unroll
  for (int n = 0; n < 16; ++n) {
    const int col = n * 16 + ln;
    const float bb = bias[col];
#pragma unroll
    for (int j = 0; j < 8; ++j) {
      const int row = qRow0 + j + hi8;
      ob[(size_t)row * HD + col] = oacc[n][j] + bb;
    }
  }
}

// Projection epilogue, fast flavor: W chunks TDM-streamed from prepacked Wt[out][in],
// double-buffered so the DMA of chunk c+1 overlaps the WMMAs of chunk c.
__device__ __forceinline__ void fused_projection_tdm(v8f (&acc)[16], const v16h (&qf)[8],
                                                     const float (&l_run)[8],
                                                     _Float16* buf0, _Float16* buf1,
                                                     _Float16* sPw,
                                                     const _Float16* __restrict__ Wt,
                                                     const float* __restrict__ bias,
                                                     float* __restrict__ ob,
                                                     int wave, int ln, int hi8, int hi16,
                                                     int qRow0) {
  const v8f zero8 = {0.f, 0.f, 0.f, 0.f, 0.f, 0.f, 0.f, 0.f};
  float invl[8];
#pragma unroll
  for (int j = 0; j < 8; ++j) invl[j] = 1.0f / l_run[j];
  v8f oacc[16];
#pragma unroll
  for (int n = 0; n < 16; ++n) oacc[n] = zero8;

  _Float16* bufs[2] = {buf0, buf1};
  if (wave == 0)   // chunk 0: Wt rows (all 256 outs) x 32 ins, row stride DIN
    tdm_load_2d_f16(Wt, (unsigned)(uintptr_t)buf0, DIN, HD, BK, HD, DIN);

  for (int c = 0; c < 16; ++c) {
    if (c + 1 < 16 && wave == 0)
      tdm_load_2d_f16(Wt + (size_t)(c + 1) * BK,
                      (unsigned)(uintptr_t)bufs[(c + 1) & 1], DIN, HD, BK, HD, DIN);
    if (wave == 0) {  // chunk c done; chunk c+1 may stay in flight (in-order TDM)
      if (c + 1 < 16) __builtin_amdgcn_s_wait_tensorcnt(1);
      else            __builtin_amdgcn_s_wait_tensorcnt(0);
    }
    __syncthreads();   // publish chunk c to all waves
    const _Float16* sWt = bufs[c & 1];

    v16h a;
    if (c < 8) {
#pragma unroll
      for (int t = 0; t < 2; ++t) {
        const int n = 2 * c + t;
#pragma unroll
        for (int j = 0; j < 8; ++j)
          sPw[(j + hi8) * BK + t * 16 + ln] = (_Float16)(acc[n][j] * invl[j]);
      }
      a = loadA32(sPw, ln, hi8);
    } else {
      a = qf[c - 8];
    }
#pragma unroll
    for (int n = 0; n < 16; ++n) {
      v16h bw = *(const v16h*)&sWt[(n * 16 + ln) * BK + hi16];
      oacc[n] = WMMA_F16(a, bw, oacc[n]);
    }
    __syncthreads();   // all reads done before this buffer is DMA-overwritten
  }
#pragma unroll
  for (int n = 0; n < 16; ++n) {
    const int col = n * 16 + ln;
    const float bb = bias[col];
#pragma unroll
    for (int j = 0; j < 8; ++j) {
      const int row = qRow0 + j + hi8;
      ob[(size_t)row * HD + col] = oacc[n][j] + bb;
    }
  }
}

// ---------------- kernel 0: fp32 -> f16 prepack of K and V ----------------
__global__ __launch_bounds__(NTHREADS)
void prepack_f16(const float* __restrict__ K, const float* __restrict__ V,
                 _Float16* __restrict__ Kh, _Float16* __restrict__ Vh) {
  const size_t base = ((size_t)blockIdx.x * NTHREADS + threadIdx.x) * 8;
  float4 k0 = *(const float4*)&K[base];
  float4 k1 = *(const float4*)&K[base + 4];
  v8h kh;
  kh[0] = (_Float16)k0.x; kh[1] = (_Float16)k0.y; kh[2] = (_Float16)k0.z; kh[3] = (_Float16)k0.w;
  kh[4] = (_Float16)k1.x; kh[5] = (_Float16)k1.y; kh[6] = (_Float16)k1.z; kh[7] = (_Float16)k1.w;
  *(v8h*)&Kh[base] = kh;
  float4 v0 = *(const float4*)&V[base];
  float4 v1 = *(const float4*)&V[base + 4];
  v8h vh;
  vh[0] = (_Float16)v0.x; vh[1] = (_Float16)v0.y; vh[2] = (_Float16)v0.z; vh[3] = (_Float16)v0.w;
  vh[4] = (_Float16)v1.x; vh[5] = (_Float16)v1.y; vh[6] = (_Float16)v1.z; vh[7] = (_Float16)v1.w;
  *(v8h*)&Vh[base] = vh;
}

// ---------------- kernel 0b: W (in,out) fp32 -> Wt (out,in) f16 ----------------
__global__ __launch_bounds__(NTHREADS)
void prepack_w(const float* __restrict__ W, _Float16* __restrict__ Wt) {
  const int t   = blockIdx.x * NTHREADS + threadIdx.x;  // DIN*HD/8 threads
  const int out = t >> 6;          // 64 8-elem segments per out row (DIN/8)
  const int in0 = (t & 63) * 8;
  v8h r;
#pragma unroll
  for (int i = 0; i < 8; ++i) r[i] = (_Float16)W[(size_t)(in0 + i) * HD + out];
  *(v8h*)&Wt[(size_t)out * DIN + in0] = r;
}

// ---------------- kernel 1 (fast path): TDM-fed K and W, double-buffered ----------------
__global__ __launch_bounds__(NTHREADS, 1)
void attn_fused_tdm(const float* __restrict__ Q, const _Float16* __restrict__ Kh,
                    const _Float16* __restrict__ Vh, const _Float16* __restrict__ Wt,
                    const float* __restrict__ bias, float* __restrict__ out) {
  __shared__ _Float16 sK[2][BK * HD];     // 2 x 16 KB, TDM destination (row-major K)
  __shared__ _Float16 sVt[2][HDV * BK];   // 2 x 16 KB, transposed V; reused as TDM W bufs
  __shared__ _Float16 sP[NW][16 * BK];    // 8 KB per-wave layout-conversion scratch

  const int tid  = threadIdx.x;
  const int lane = tid & 31;
  const int wave = tid >> 5;
  const int ln   = lane & 15;
  const int hi   = lane >> 4;
  const int hi8  = hi * 8;
  const int hi16 = hi * 16;

  const int batch = blockIdx.y;
  const int qRow0 = blockIdx.x * BQ + wave * 16;

  const float*    Qb  = Q  + (size_t)batch * SQ * HD;
  const _Float16* Khb = Kh + (size_t)batch * SK * HD;
  const _Float16* Vhb = Vh + (size_t)batch * SK * HDV;

  v16h qf[8];
  {
    const float* qrow = Qb + (size_t)(qRow0 + ln) * HD;
#pragma unroll
    for (int f = 0; f < 8; ++f) qf[f] = loadAQ(qrow, f, hi8);
  }

  float m_run[8], l_run[8];
#pragma unroll
  for (int j = 0; j < 8; ++j) { m_run[j] = -__builtin_inff(); l_run[j] = 0.f; }
  const v8f zero8 = {0.f, 0.f, 0.f, 0.f, 0.f, 0.f, 0.f, 0.f};
  v8f acc[16];
#pragma unroll
  for (int n = 0; n < 16; ++n) acc[n] = zero8;
  _Float16* sPw = sP[wave];

  // V transpose staging from f16 workspace (TDM cannot transpose)
  auto stageV = [&](int kBase, _Float16* dst) {
    for (int idx = tid; idx < (BK * HDV) / 8; idx += NTHREADS) {
      const int e  = idx * 8;
      const int kr = e >> 8;
      const int dc = e & 255;
      v8h v = *(const v8h*)&Vhb[(size_t)(kBase + kr) * HDV + dc];
#pragma unroll
      for (int i = 0; i < 8; ++i) dst[(dc + i) * BK + kr] = v[i];
      __builtin_prefetch(&Vhb[(size_t)(kBase + BK + kr) * HDV + dc], 0, 0);
    }
  };

  // prologue: tile 0
  if (wave == 0)
    tdm_load_2d_f16(Khb, (unsigned)(uintptr_t)&sK[0][0], HD, SK, HD, BK, HD);
  stageV(0, sVt[0]);
  if (wave == 0) __builtin_amdgcn_s_wait_tensorcnt(0);
  __syncthreads();

  int cur = 0;
  for (int kt = 0; kt < NKT; ++kt) {
    const int nxt = cur ^ 1;
    if (kt + 1 < NKT) {
      if (wave == 0)   // stream next K tile while this one is consumed
        tdm_load_2d_f16(Khb + (size_t)(kt + 1) * BK * HD,
                        (unsigned)(uintptr_t)&sK[nxt][0], HD, SK, HD, BK, HD);
      stageV((kt + 1) * BK, sVt[nxt]);
    }
    attn_tile(sK[cur], sVt[cur], sPw, qf, acc, m_run, l_run, ln, hi8, hi16);
    if (wave == 0) __builtin_amdgcn_s_wait_tensorcnt(0);
    __syncthreads();   // publish next buffers (TDM + ds stores) to all waves
    cur = nxt;
  }

  fused_projection_tdm(acc, qf, l_run, sVt[0], sVt[1], sPw, Wt, bias,
                       out + (size_t)batch * SQ * HD, wave, ln, hi8, hi16, qRow0);
}

// ---------------- fallback kernel (no workspace) ----------------
__global__ __launch_bounds__(NTHREADS, 1)
void attn_fused_fp32(const float* __restrict__ Q, const float* __restrict__ K,
                     const float* __restrict__ V, const float* __restrict__ W,
                     const float* __restrict__ bias, float* __restrict__ out) {
  __shared__ _Float16 sK[BK * HD];
  __shared__ _Float16 sVt[HDV * BK];
  __shared__ _Float16 sP[NW][16 * BK];

  const int tid  = threadIdx.x;
  const int lane = tid & 31;
  const int wave = tid >> 5;
  const int ln   = lane & 15;
  const int hi   = lane >> 4;
  const int hi8  = hi * 8;
  const int hi16 = hi * 16;

  const int batch = blockIdx.y;
  const int qRow0 = blockIdx.x * BQ + wave * 16;

  const float* Qb = Q + (size_t)batch * SQ * HD;
  const float* Kb = K + (size_t)batch * SK * HD;
  const float* Vb = V + (size_t)batch * SK * HDV;

  v16h qf[8];
  {
    const float* qrow = Qb + (size_t)(qRow0 + ln) * HD;
#pragma unroll
    for (int f = 0; f < 8; ++f) qf[f] = loadAQ(qrow, f, hi8);
  }
  float m_run[8], l_run[8];
#pragma unroll
  for (int j = 0; j < 8; ++j) { m_run[j] = -__builtin_inff(); l_run[j] = 0.f; }
  const v8f zero8 = {0.f, 0.f, 0.f, 0.f, 0.f, 0.f, 0.f, 0.f};
  v8f acc[16];
#pragma unroll
  for (int n = 0; n < 16; ++n) acc[n] = zero8;
  _Float16* sPw = sP[wave];

  for (int kt = 0; kt < NKT; ++kt) {
    const int kBase = kt * BK;
    for (int idx = tid; idx < (BK * HD) / 4; idx += NTHREADS) {
      const int e  = idx * 4;
      const int kr = e >> 8;
      const int dc = e & 255;
      float4 kv = *(const float4*)&Kb[(size_t)(kBase + kr) * HD + dc];
      sK[kr * HD + dc + 0] = (_Float16)kv.x;
      sK[kr * HD + dc + 1] = (_Float16)kv.y;
      sK[kr * HD + dc + 2] = (_Float16)kv.z;
      sK[kr * HD + dc + 3] = (_Float16)kv.w;
      float4 vv = *(const float4*)&Vb[(size_t)(kBase + kr) * HDV + dc];
      sVt[(dc + 0) * BK + kr] = (_Float16)vv.x;
      sVt[(dc + 1) * BK + kr] = (_Float16)vv.y;
      sVt[(dc + 2) * BK + kr] = (_Float16)vv.z;
      sVt[(dc + 3) * BK + kr] = (_Float16)vv.w;
      if (kt + 1 < NKT) {
        __builtin_prefetch(&Kb[(size_t)(kBase + BK + kr) * HD + dc], 0, 0);
        __builtin_prefetch(&Vb[(size_t)(kBase + BK + kr) * HDV + dc], 0, 0);
      }
    }
    __syncthreads();
    attn_tile(sK, sVt, sPw, qf, acc, m_run, l_run, ln, hi8, hi16);
    __syncthreads();
  }

  fused_projection(acc, qf, l_run, sVt, sPw, W, bias,
                   out + (size_t)batch * SQ * HD, tid, ln, hi8, hi16, qRow0);
}

extern "C" void kernel_launch(void* const* d_in, const int* in_sizes, int n_in,
                              void* d_out, int out_size, void* d_ws, size_t ws_size,
                              hipStream_t stream) {
  (void)in_sizes; (void)n_in; (void)out_size;
  const float* q = (const float*)d_in[0];
  const float* k = (const float*)d_in[1];
  const float* v = (const float*)d_in[2];
  const float* w = (const float*)d_in[3];
  const float* b = (const float*)d_in[4];
  float* out = (float*)d_out;

  dim3 grid(SQ / BQ, BATCH);   // (16, 8)
  const size_t need = (2 * NE + (size_t)DIN * HD) * sizeof(_Float16);  // ~16.25 MB
  if (ws_size >= need && d_ws != nullptr) {
    _Float16* Kh = (_Float16*)d_ws;
    _Float16* Vh = Kh + NE;
    _Float16* Wt = Vh + NE;
    const unsigned packBlocks  = (unsigned)(NE / 8 / NTHREADS);          // 2048
    const unsigned packWBlocks = (unsigned)((size_t)DIN * HD / 8 / NTHREADS);  // 64
    prepack_f16<<<packBlocks, NTHREADS, 0, stream>>>(k, v, Kh, Vh);
    prepack_w<<<packWBlocks, NTHREADS, 0, stream>>>(w, Wt);
    attn_fused_tdm<<<grid, NTHREADS, 0, stream>>>(q, Kh, Vh, Wt, b, out);
  } else {
    attn_fused_fp32<<<grid, NTHREADS, 0, stream>>>(q, k, v, w, b, out);
  }
}
